// MultiScale_Points_Volume_Encoder_39986145526179
// MI455X (gfx1250) — compile-verified, matrix operand order checked by reading, not run on hardware
//
#include <hip/hip_runtime.h>
#include <hip/hip_bf16.h>
#include <math.h>

// ---------------------------------------------------------------------------
// MultiScale Points + Volume Encoder for MI455X (gfx1250, wave32, WMMA).
// B=2, N=8192, CIN=32, COUT=64, K=16 neighbors, SCALES={32,16,8}.
// GEMM-like stages use v_wmma_f32_16x16x32_f16.  LDS tiles are staged as
// f16 pre-swizzled into the WMMA register layout, so the inner loop is pure
// ds_load_b128 -> v_wmma with zero VALU converts.
// ---------------------------------------------------------------------------

typedef __attribute__((ext_vector_type(16))) _Float16 v16h;
typedef __attribute__((ext_vector_type(8)))  float    v8f;

#define NB   2
#define NP   8192
#define PTS  16384L        // NB*NP
#define ROWS 262144L       // NB*NP*16
#define CIN_ 32
#define COUT_ 64

// Swizzle helper: element (row m / col n, k) -> [k/32][m+16*hi][j]
// hi = (k>>3)&1  (which half-wave lane group holds this k)
// j  = (k&7) + 8*((k>>4)&1)   (position inside the v16h)
__device__ inline void swz(int k, int& kblk, int& hi, int& j) {
    kblk = k >> 5;
    int kk = k & 31;
    hi = (kk >> 3) & 1;
    j  = (kk & 7) + (((kk >> 4) & 1) << 3);
}

// ---------------------------------------------------------------------------
// Generic GEMM + affine + leaky-relu (+optional residual) kernel.
// Y[row, ycol0+n] = lrelu((X[row,:K] @ W[:,n] + b[n]) * g[n] + be[n] (+res))
// 8 waves/block; each wave computes a 16 x (16*NT) strip: NT accumulators
// share a single A operand per 32-wide K step.  KP in {32,64} compile-time.
// ---------------------------------------------------------------------------
template <int KP, int NT>
__global__ void __launch_bounds__(256) mlp_wmma_kernel(
    const float* __restrict__ X, int ldx, long M, int Kreal,
    const float* __restrict__ Wm, int Nw,
    const float* __restrict__ bias, const float* __restrict__ beta,
    const float* __restrict__ gain,
    float slope, const float* __restrict__ resid, int ldr,
    float* __restrict__ Y, int ldy, int ycol0)
{
    constexpr int KB = KP / 32;
    __shared__ alignas(32) _Float16 hA[8][KB][32][16];
    __shared__ alignas(32) _Float16 hW[NT][KB][32][16];

    int tid  = threadIdx.x;
    int wave = tid >> 5;
    int lane = tid & 31;
    int ng0  = blockIdx.y * NT * 16;         // first output column of block

    // stage W strip (NT tiles of 16 cols), f16-swizzled
    for (int t = tid; t < NT * 16 * KP; t += 256) {
        int n = t / KP, k = t % KP;
        int ng = ng0 + n;
        float v = (ng < Nw && k < Kreal) ? Wm[(long)k * Nw + ng] : 0.0f;
        int kblk, hi, j; swz(k, kblk, hi, j);
        hW[n >> 4][kblk][(n & 15) + (hi << 4)][j] = (_Float16)v;
    }
    // stage A tile (per wave), f16-swizzled
    long mtile = (long)blockIdx.x * 8 + wave;
    long r0 = mtile * 16;
    for (int t = lane; t < 16 * KP; t += 32) {
        int m = t / KP, k = t % KP;
        long row = r0 + m;
        float v = (row < M && k < Kreal) ? X[row * (long)ldx + k] : 0.0f;
        int kblk, hi, j; swz(k, kblk, hi, j);
        hA[wave][kblk][m + (hi << 4)][j] = (_Float16)v;
    }
    __syncthreads();

    v8f acc[NT];
#pragma unroll
    for (int nt = 0; nt < NT; nt++) acc[nt] = {};

#pragma unroll
    for (int kblk = 0; kblk < KB; kblk++) {
        v16h a = *(const v16h*)(&hA[wave][kblk][lane][0]);
#pragma unroll
        for (int nt = 0; nt < NT; nt++) {
            v16h b = *(const v16h*)(&hW[nt][kblk][lane][0]);
            acc[nt] = __builtin_amdgcn_wmma_f32_16x16x32_f16(
                          false, a, false, b, (short)0, acc[nt], false, false);
        }
    }

    int nl = lane & 15;
    int mhi = (lane >> 4) * 8;
#pragma unroll
    for (int nt = 0; nt < NT; nt++) {
        int ng = ng0 + nt * 16 + nl;
        if (ng < Nw) {
            float bb = bias ? bias[ng] : 0.0f;
            float gg = gain ? gain[ng] : 1.0f;
            float ee = beta ? beta[ng] : 0.0f;
#pragma unroll
            for (int rr = 0; rr < 8; rr++) {
                long row = r0 + mhi + rr;
                if (row < M) {
                    float y = (acc[nt][rr] + bb) * gg + ee;
                    if (resid) y += resid[row * (long)ldr + ng];
                    y = (y >= 0.0f) ? y : slope * y;
                    Y[row * (long)ldy + ycol0 + ng] = y;
                }
            }
        }
    }
}

// ---------------------------------------------------------------------------
// Implicit-GEMM 3x3x3 conv (pad 1) with affine + leaky-relu, via WMMA.
// In  : (B, Ci, r^3)   Kw : (Co, Ci, 27)   Out : (B, Co, r^3)
// Each wave: 16 output voxels x 16 output channels; 27 taps x CBS WMMAs.
// ---------------------------------------------------------------------------
template <int CBS>   // Ci = 32*CBS
__global__ void __launch_bounds__(256) conv3d_wmma_kernel(
    const float* __restrict__ In, const float* __restrict__ Kw,
    const float* __restrict__ bias, const float* __restrict__ beta,
    const float* __restrict__ gain, float slope,
    float* __restrict__ Out, int r, int Co, int Bn)
{
    __shared__ alignas(32) _Float16 hA[8][32][16];
    __shared__ alignas(32) _Float16 hW[8][32][16];

    const int Ci = 32 * CBS;
    int tid = threadIdx.x, wave = tid >> 5, lane = tid & 31;
    long r3 = (long)r * r * r;
    int ptpb = (int)(r3 / 16);
    int coT  = Co / 16;
    long waveId = (long)blockIdx.x * 8 + wave;
    long total  = (long)Bn * ptpb * coT;
    bool active = waveId < total;

    int b = 0, pt = 0, ct = 0;
    if (active) {
        b = (int)(waveId / ((long)ptpb * coT));
        long rem = waveId % ((long)ptpb * coT);
        pt = (int)(rem / coT);
        ct = (int)(rem % coT);
    }
    int p0 = pt * 16, co0 = ct * 16;
    v8f acc = {};

    for (int t = 0; t < 27; t++) {
        int dz = t / 9 - 1, dy = (t / 3) % 3 - 1, dx = t % 3 - 1;
#pragma unroll
        for (int cb = 0; cb < CBS; cb++) {
            if (active) {
                for (int tt = lane; tt < 512; tt += 32) {
                    int m = tt >> 5, c = tt & 31;
                    int p = p0 + m;
                    int z = p / (r * r), rem2 = p % (r * r);
                    int y = rem2 / r, x = rem2 % r;
                    int zz = z + dz, yy = y + dy, xx = x + dx;
                    float v = 0.0f;
                    if (zz >= 0 && zz < r && yy >= 0 && yy < r && xx >= 0 && xx < r)
                        v = In[(((long)(b * Ci + cb * 32 + c) * r + zz) * r + yy) * r + xx];
                    int kblk, hi, j; swz(c, kblk, hi, j);
                    hA[wave][m + (hi << 4)][j] = (_Float16)v;
                }
                for (int tt = lane; tt < 512; tt += 32) {
                    int n = tt >> 5, c = tt & 31;
                    float v = Kw[((long)(co0 + n) * Ci + cb * 32 + c) * 27 + t];
                    int kblk, hi, j; swz(c, kblk, hi, j);
                    hW[wave][n + (hi << 4)][j] = (_Float16)v;
                }
            }
            __syncthreads();
            if (active) {
                v16h a = *(const v16h*)(&hA[wave][lane][0]);
                v16h w = *(const v16h*)(&hW[wave][lane][0]);
                acc = __builtin_amdgcn_wmma_f32_16x16x32_f16(
                          false, a, false, w, (short)0, acc, false, false);
            }
            __syncthreads();
        }
    }

    if (active) {
        int nl = lane & 15;
        int co = co0 + nl;
        float bb = bias[co], gg = gain[co], ee = beta[co];
        int mhi = (lane >> 4) * 8;
#pragma unroll
        for (int rr = 0; rr < 8; rr++) {
            int p = p0 + mhi + rr;
            float y = (acc[rr] + bb) * gg + ee;
            y = (y >= 0.0f) ? y : slope * y;
            Out[(long)(b * Co + co) * r3 + p] = y;
        }
    }
}

// ---------------------------------------------------------------------------
// KNN: per-thread query, 512-point LDS candidate tiles, register top-16.
// Next tile is prefetched (global_prefetch_b8) while current is scanned.
// ---------------------------------------------------------------------------
__global__ void __launch_bounds__(128) knn_kernel(
    const float* __restrict__ pos, int* __restrict__ out, int Npts)
{
    __shared__ float sp[512 * 3];
    int tid = threadIdx.x;
    int bpb = Npts / 128;
    int b = blockIdx.x / bpb;
    int n = (blockIdx.x % bpb) * 128 + tid;
    const float* base = pos + (long)b * Npts * 3;
    float qx = base[n * 3 + 0], qy = base[n * 3 + 1], qz = base[n * 3 + 2];

    float bd[16]; int bi[16];
#pragma unroll
    for (int q = 0; q < 16; q++) { bd[q] = 3.4e38f; bi[q] = 0; }

    for (int c0 = 0; c0 < Npts; c0 += 512) {
        for (int t = tid; t < 512; t += 128) {
            sp[t * 3 + 0] = base[(c0 + t) * 3 + 0];
            sp[t * 3 + 1] = base[(c0 + t) * 3 + 1];
            sp[t * 3 + 2] = base[(c0 + t) * 3 + 2];
        }
        __syncthreads();
        // prefetch next tile (6KB = 24 x 256B) while scanning this one
        int nxt = c0 + 512;
        if (nxt < Npts && tid < 24)
            __builtin_prefetch(base + (long)nxt * 3 + tid * 64, 0, 1);
        for (int j = 0; j < 512; j++) {
            float dx = sp[j * 3 + 0] - qx;
            float dy = sp[j * 3 + 1] - qy;
            float dz = sp[j * 3 + 2] - qz;
            float d2 = dx * dx + dy * dy + dz * dz;
            if (d2 < bd[15]) {
                int cid = c0 + j;
                bool placed = false;
#pragma unroll
                for (int q = 15; q >= 1; q--) {
                    if (!placed) {
                        if (bd[q - 1] > d2) { bd[q] = bd[q - 1]; bi[q] = bi[q - 1]; }
                        else { bd[q] = d2; bi[q] = cid; placed = true; }
                    }
                }
                if (!placed) { bd[0] = d2; bi[0] = cid; }
            }
        }
        __syncthreads();
    }
    int* o = out + ((long)b * Npts + n) * 16;
#pragma unroll
    for (int q = 0; q < 16; q++) o[q] = bi[q];
}

// ---------------------------------------------------------------------------
// rel = [pos_i, pos_j, diff, dist]  -> A rows (stride 64), cols 0..9
// ---------------------------------------------------------------------------
__global__ void build_rel_kernel(const float* __restrict__ pos,
                                 const int* __restrict__ idx,
                                 float* __restrict__ A, int Npts, long total)
{
    long i = (long)blockIdx.x * 256 + threadIdx.x;
    if (i >= total) return;
    long pn = i / 16;
    int b = (int)(pn / Npts);
    int j = idx[i];
    const float* pi = pos + pn * 3;
    const float* pj = pos + ((long)b * Npts + j) * 3;
    float dx = pj[0] - pi[0], dy = pj[1] - pi[1], dz = pj[2] - pi[2];
    float* o = A + i * 64;
    o[0] = pi[0]; o[1] = pi[1]; o[2] = pi[2];
    o[3] = pj[0]; o[4] = pj[1]; o[5] = pj[2];
    o[6] = dx;    o[7] = dy;    o[8] = dz;
    o[9] = sqrtf(dx * dx + dy * dy + dz * dz);
}

__global__ void gather_xj_kernel(const float* __restrict__ h, int C, int ldh,
                                 const int* __restrict__ idx,
                                 float* __restrict__ LOC, int col0,
                                 int Npts, long total)
{
    long i = (long)blockIdx.x * 256 + threadIdx.x;
    if (i >= total) return;
    long pn = i / 16;
    int b = (int)(pn / Npts);
    int j = idx[i];
    const float* src = h + ((long)b * Npts + j) * ldh;
    float* dst = LOC + i * 64 + col0;
    for (int c = 0; c < C; c++) dst[c] = src[c];
}

__global__ void build_edge_kernel(const float* __restrict__ h, int C, int ldh,
                                  const int* __restrict__ idx,
                                  float* __restrict__ A, int Npts, long total)
{
    long i = (long)blockIdx.x * 256 + threadIdx.x;
    if (i >= total) return;
    long pn = i / 16;
    int b = (int)(pn / Npts);
    int j = idx[i];
    const float* xi = h + pn * (long)ldh;
    const float* xj = h + ((long)b * Npts + j) * ldh;
    float* o = A + i * 64;
    for (int c = 0; c < C; c++) {
        float a = xi[c], bb = xj[c];
        o[c] = a; o[C + c] = bb; o[2 * C + c] = bb - a;
    }
}

// softmax over 16 neighbors per (point, channel); att-weighted + raw sums.
__global__ void reduce_att_kernel(const float* __restrict__ L,
                                  const float* __restrict__ LOC, int twoCh,
                                  float* __restrict__ SATT,
                                  float* __restrict__ SRAW, long P)
{
    long i = (long)blockIdx.x * 256 + threadIdx.x;
    if (i >= P * twoCh) return;
    int c = (int)(i % twoCh);
    long p = i / twoCh;
    long base = p * 16;
    float mx = -3.4e38f;
    for (int k = 0; k < 16; k++) mx = fmaxf(mx, L[(base + k) * 64 + c]);
    float s = 0.0f, wv = 0.0f, ls = 0.0f;
    for (int k = 0; k < 16; k++) {
        float e  = expf(L[(base + k) * 64 + c] - mx);
        float lv = LOC[(base + k) * 64 + c];
        s += e; wv += e * lv; ls += lv;
    }
    SATT[p * 64 + c] = wv / s;
    if (SRAW) SRAW[p * 64 + c] = ls;
}

// ---------------------------------------------------------------------------
// Voxelization: per-batch mean + radius, then atomic scatter-mean.
// ---------------------------------------------------------------------------
__global__ void __launch_bounds__(256) vox_stats_kernel(
    const float* __restrict__ LC, float* __restrict__ stats, int Npts)
{
    int b = blockIdx.x, tid = threadIdx.x;
    __shared__ float sx[256], sy[256], sz[256];
    __shared__ float mean[3];
    float ax = 0, ay = 0, az = 0;
    for (int n = tid; n < Npts; n += 256) {
        const float* p = LC + ((long)b * Npts + n) * 3;
        ax += p[0]; ay += p[1]; az += p[2];
    }
    sx[tid] = ax; sy[tid] = ay; sz[tid] = az;
    __syncthreads();
    for (int off = 128; off > 0; off >>= 1) {
        if (tid < off) { sx[tid] += sx[tid + off]; sy[tid] += sy[tid + off]; sz[tid] += sz[tid + off]; }
        __syncthreads();
    }
    if (tid == 0) { mean[0] = sx[0] / Npts; mean[1] = sy[0] / Npts; mean[2] = sz[0] / Npts; }
    __syncthreads();
    float mx = 0;
    for (int n = tid; n < Npts; n += 256) {
        const float* p = LC + ((long)b * Npts + n) * 3;
        float dx = p[0] - mean[0], dy = p[1] - mean[1], dz = p[2] - mean[2];
        mx = fmaxf(mx, sqrtf(dx * dx + dy * dy + dz * dz));
    }
    sx[tid] = mx;
    __syncthreads();
    for (int off = 128; off > 0; off >>= 1) {
        if (tid < off) sx[tid] = fmaxf(sx[tid], sx[tid + off]);
        __syncthreads();
    }
    if (tid == 0) {
        stats[b * 4 + 0] = mean[0]; stats[b * 4 + 1] = mean[1];
        stats[b * 4 + 2] = mean[2]; stats[b * 4 + 3] = sx[0];
    }
}

__global__ void fill0_kernel(float* __restrict__ p, long n)
{
    long i = (long)blockIdx.x * 256 + threadIdx.x;
    if (i < n) p[i] = 0.0f;
}

__global__ void vox_accum_kernel(const float* __restrict__ LC,
                                 const float* __restrict__ Pf,
                                 const float* __restrict__ stats, int r,
                                 float* __restrict__ sums, float* __restrict__ cnt,
                                 float* __restrict__ nc0, int Npts, long total)
{
    long i = (long)blockIdx.x * 256 + threadIdx.x;
    if (i >= total) return;
    int b = (int)(i / Npts);
    const float* st = stats + b * 4;
    float denom = st[3] * 2.0f + 1e-6f;
    float v[3];
#pragma unroll
    for (int d = 0; d < 3; d++) {
        float nc = (LC[i * 3 + d] - st[d]) / denom + 0.5f;
        nc = nc * (float)r;
        nc = fminf(fmaxf(nc, 0.0f), (float)(r - 1));
        v[d] = nc;
    }
    if (nc0) { nc0[i * 3 + 0] = v[0]; nc0[i * 3 + 1] = v[1]; nc0[i * 3 + 2] = v[2]; }
    int vx = (int)rintf(v[0]), vy = (int)rintf(v[1]), vz = (int)rintf(v[2]);
    long r3 = (long)r * r * r;
    long flat = (long)b * r3 + ((long)vx * r + vy) * r + vz;
    atomicAdd(&cnt[flat], 1.0f);
    for (int c = 0; c < CIN_; c++)
        atomicAdd(&sums[flat * CIN_ + c], Pf[i * CIN_ + c]);
}

__global__ void vox_final_kernel(const float* __restrict__ sums,
                                 const float* __restrict__ cnt,
                                 float* __restrict__ gin, int r, long total)
{
    long i = (long)blockIdx.x * 256 + threadIdx.x;
    if (i >= total) return;
    long r3 = (long)r * r * r;
    int b = (int)(i / r3);
    long vcell = i % r3;
    float d = fmaxf(cnt[i], 1.0f);
    for (int c = 0; c < CIN_; c++)
        gin[(long)(b * CIN_ + c) * r3 + vcell] = sums[i * CIN_ + c] / d;
}

// ---------------------------------------------------------------------------
// Fused trilinear upsample (16->32, 8->32) + gates + softmax + weighted fuse.
// 64 threads per voxel (one per channel), 4 voxels per block.
// ---------------------------------------------------------------------------
__device__ inline float tri3(const float* __restrict__ G, int b, int c, int I,
                             long I3, int D, int H, int W)
{
    float sc = (float)(I - 1) / 31.0f;
    float pd = D * sc, ph = H * sc, pw = W * sc;
    int ld = min((int)floorf(pd), I - 2); float wd = pd - ld;
    int lh = min((int)floorf(ph), I - 2); float wh = ph - lh;
    int lw = min((int)floorf(pw), I - 2); float ww = pw - lw;
    const float* base = G + (long)(b * 64 + c) * I3;
    float v = 0.0f;
#pragma unroll
    for (int dz = 0; dz < 2; dz++)
#pragma unroll
        for (int dy = 0; dy < 2; dy++)
#pragma unroll
            for (int dx = 0; dx < 2; dx++) {
                float w = (dz ? wd : 1.0f - wd) * (dy ? wh : 1.0f - wh) * (dx ? ww : 1.0f - ww);
                v += w * base[((long)(ld + dz) * I + (lh + dy)) * I + (lw + dx)];
            }
    return v;
}

__global__ void __launch_bounds__(256) upgate_kernel(
    const float* __restrict__ G32, const float* __restrict__ G16,
    const float* __restrict__ G8,
    const float* gw0, const float* gb0, const float* gbe0, const float* gg0,
    const float* gw1, const float* gb1, const float* gbe1, const float* gg1,
    const float* gw2, const float* gb2, const float* gbe2, const float* gg2,
    float* __restrict__ FIN)
{
    int tid = threadIdx.x;
    int c = tid & 63, pl = tid >> 6;
    long gpos = (long)blockIdx.x * 4 + pl;            // 0 .. B*32768-1
    int b = (int)(gpos >> 15);
    int m = (int)(gpos & 32767);
    int D = m >> 10, H = (m >> 5) & 31, Wc = m & 31;

    float f[3];
    f[0] = G32[((long)(b * 64 + c) << 15) + m];
    f[1] = tri3(G16, b, c, 16, 4096, D, H, Wc);
    f[2] = tri3(G8,  b, c,  8,  512, D, H, Wc);

    const float* gw[3]  = { gw0, gw1, gw2 };
    const float* gb[3]  = { gb0, gb1, gb2 };
    const float* gbe[3] = { gbe0, gbe1, gbe2 };
    const float* gg[3]  = { gg0, gg1, gg2 };

    __shared__ float sred[4][9][64];
    __shared__ float wsh[4][3];
#pragma unroll
    for (int i = 0; i < 3; i++)
#pragma unroll
        for (int o = 0; o < 3; o++)
            sred[pl][i * 3 + o][c] = f[i] * gw[i][c * 3 + o];
    __syncthreads();
    for (int off = 32; off > 0; off >>= 1) {
        if (c < off)
            for (int io = 0; io < 9; io++)
                sred[pl][io][c] += sred[pl][io][c + off];
        __syncthreads();
    }
    if (c == 0) {
        float gs[3];
#pragma unroll
        for (int o = 0; o < 3; o++) {
            float acc = 0.0f;
#pragma unroll
            for (int i = 0; i < 3; i++) {
                float y = sred[pl][i * 3 + o][0] + gb[i][o];
                y = y * gg[i][o] + gbe[i][o];
                y = (y >= 0.0f) ? y : 0.2f * y;
                acc += 1.0f / (1.0f + expf(-y));
            }
            gs[o] = acc;
        }
        float mx = fmaxf(gs[0], fmaxf(gs[1], gs[2]));
        float e0 = expf(gs[0] - mx), e1 = expf(gs[1] - mx), e2 = expf(gs[2] - mx);
        float s = e0 + e1 + e2;
        wsh[pl][0] = e0 / s; wsh[pl][1] = e1 / s; wsh[pl][2] = e2 / s;
    }
    __syncthreads();
    float fu = wsh[pl][0] * f[0] + wsh[pl][1] * f[1] + wsh[pl][2] * f[2];
    FIN[gpos * 64 + c] = fu;
}

// trilinear devoxelize of fused grid back onto points; OUT is (B,64,N)
__global__ void __launch_bounds__(256) devox_kernel(
    const float* __restrict__ FOUT, const float* __restrict__ NC0,
    float* __restrict__ OUT)
{
    int tid = threadIdx.x;
    int c = tid & 63, pl = tid >> 6;
    long gp = (long)blockIdx.x * 4 + pl;              // 0 .. B*N-1
    int b = (int)(gp >> 13);
    int n = (int)(gp & 8191);
    const float* nc = NC0 + gp * 3;
    float fx = nc[0], fy = nc[1], fz = nc[2];
    int lx = min(max((int)floorf(fx), 0), 30);
    int ly = min(max((int)floorf(fy), 0), 30);
    int lz = min(max((int)floorf(fz), 0), 30);
    float ax = fx - lx, ay = fy - ly, az = fz - lz;
    const float* base = FOUT + ((long)b << 15) * 64;
    float v = 0.0f;
#pragma unroll
    for (int dx = 0; dx < 2; dx++)
#pragma unroll
        for (int dy = 0; dy < 2; dy++)
#pragma unroll
            for (int dz = 0; dz < 2; dz++) {
                float w = (dx ? ax : 1.0f - ax) * (dy ? ay : 1.0f - ay) * (dz ? az : 1.0f - az);
                long idx = ((long)(lx + dx) * 32 + (ly + dy)) * 32 + (lz + dz);
                v += w * base[idx * 64 + c];
            }
    OUT[(long)(b * 64 + c) * 8192 + n] = v;
}

// ---------------------------------------------------------------------------
// Host orchestration
// ---------------------------------------------------------------------------
static inline void launch_mlp(hipStream_t s, const float* X, int ldx, long M,
                              int Kreal, int Kp, const float* Wm, int Nw,
                              const float* b, const float* be, const float* g,
                              float slope, const float* resid, int ldr,
                              float* Y, int ldy, int ycol0)
{
    long mt = (M + 15) / 16;
    unsigned gx = (unsigned)((mt + 7) / 8);
    int NT = (Nw >= 64) ? 4 : (Nw >= 32) ? 2 : 1;
    unsigned gy = (unsigned)((Nw + NT * 16 - 1) / (NT * 16));
    dim3 grid(gx, gy);
    if (Kp == 32) {
        if (NT == 4)      mlp_wmma_kernel<32, 4><<<grid, 256, 0, s>>>(X, ldx, M, Kreal, Wm, Nw, b, be, g, slope, resid, ldr, Y, ldy, ycol0);
        else if (NT == 2) mlp_wmma_kernel<32, 2><<<grid, 256, 0, s>>>(X, ldx, M, Kreal, Wm, Nw, b, be, g, slope, resid, ldr, Y, ldy, ycol0);
        else              mlp_wmma_kernel<32, 1><<<grid, 256, 0, s>>>(X, ldx, M, Kreal, Wm, Nw, b, be, g, slope, resid, ldr, Y, ldy, ycol0);
    } else {
        if (NT == 4)      mlp_wmma_kernel<64, 4><<<grid, 256, 0, s>>>(X, ldx, M, Kreal, Wm, Nw, b, be, g, slope, resid, ldr, Y, ldy, ycol0);
        else if (NT == 2) mlp_wmma_kernel<64, 2><<<grid, 256, 0, s>>>(X, ldx, M, Kreal, Wm, Nw, b, be, g, slope, resid, ldr, Y, ldy, ycol0);
        else              mlp_wmma_kernel<64, 1><<<grid, 256, 0, s>>>(X, ldx, M, Kreal, Wm, Nw, b, be, g, slope, resid, ldr, Y, ldy, ycol0);
    }
}

extern "C" void kernel_launch(void* const* d_in, const int* in_sizes, int n_in,
                              void* d_out, int out_size, void* d_ws, size_t ws_size,
                              hipStream_t stream)
{
    (void)in_sizes; (void)n_in; (void)out_size; (void)ws_size;
    const float* Pf = (const float*)d_in[0];   // (B,N,32)
    const float* GC = (const float*)d_in[1];   // (B,N,3) global_coord
    const float* LC = (const float*)d_in[2];   // (B,N,3) local_coord
    auto P = [&](int i) -> const float* { return (const float*)d_in[i]; };

    // --- params (JAX sorted-key tree flatten order; lin: W,b,be,g) ---
    // 3..26: convs[s].{c1,c2}.{K,b,be,g}
    // 27..30: fuse.{W,b,be,g}   31..42: gates[i].{W,b,be,g}
    // 43: lfa1.att_W  44..47 dgcnn  48..51 enc  52..55 post  56..59 raw
    // 60: lfa2.att_W  61..64 dgcnn  65..68 enc  69..72 post
    // 73..76 mlp1  77..80 mlp2  81..84 shortcut

    float* F  = (float*)d_ws;
    int*  IDX = (int*)d_ws;
    long off = ROWS;                       // idx ints occupy first 262144 slots
    float* SHORTB = F + off; off += PTS * 64;
    float* H1     = F + off; off += PTS * 8;
    float* H2     = F + off; off += PTS * 16;
    float* H3     = F + off; off += PTS * 32;
    float* SATT   = F + off; off += PTS * 64;
    float* SRAW   = F + off; off += PTS * 64;
    float* ABUF   = F + off; off += ROWS * 64;   // rel/edge AND attention logits
    float* LOC    = F + off; off += ROWS * 64;
    float* STATS  = F + off; off += 16;
    float* NC0    = F + off; off += PTS * 3;
    float* VSUM   = F + off; off += (long)NB * 32768 * CIN_;
    float* VCNT   = F + off; off += (long)NB * 32768;
    float* GIN    = F + off; off += (long)NB * CIN_ * 32768;
    float* GTMP   = F + off; off += (long)NB * COUT_ * 32768;
    float* G32    = F + off; off += (long)NB * COUT_ * 32768;
    float* G16    = F + off; off += (long)NB * COUT_ * 4096;
    float* G8     = F + off; off += (long)NB * COUT_ * 512;
    float* FIN    = F + off; off += (long)NB * 32768 * 64;
    float* FOUT   = F + off; off += (long)NB * 32768 * 64;

    float* OUT_FP  = (float*)d_out;              // (B,N,64)
    float* OUT_REC = OUT_FP + PTS * 64;          // (B,N,64)
    float* OUT_DVX = OUT_REC + PTS * 64;         // (B,64,N)

    // ======================= points branch =======================
    knn_kernel<<<(NB * NP) / 128, 128, 0, stream>>>(GC, IDX, NP);

    // shortcut (no act) and mlp1 (lrelu 0.2)
    launch_mlp(stream, Pf, 32, PTS, 32, 32, P(81), 64, P(82), P(83), P(84),
               1.0f, nullptr, 0, SHORTB, 64, 0);
    launch_mlp(stream, Pf, 32, PTS, 32, 32, P(73), 8, P(74), P(75), P(76),
               0.2f, nullptr, 0, H1, 8, 0);

    dim3 rgrid((unsigned)((ROWS + 255) / 256));

    // ---- LFA1 (ch=16, in C=8) ----
    build_rel_kernel<<<rgrid, 256, 0, stream>>>(GC, IDX, ABUF, NP, ROWS);
    launch_mlp(stream, ABUF, 64, ROWS, 10, 32, P(48), 8, P(49), P(50), P(51),
               0.2f, nullptr, 0, LOC, 64, 24);                       // lse -> [24,32)
    gather_xj_kernel<<<rgrid, 256, 0, stream>>>(H1, 8, 8, IDX, LOC, 16, NP, ROWS);
    build_edge_kernel<<<rgrid, 256, 0, stream>>>(H1, 8, 8, IDX, ABUF, NP, ROWS);
    launch_mlp(stream, ABUF, 64, ROWS, 24, 32, P(44), 16, P(45), P(46), P(47),
               0.2f, nullptr, 0, LOC, 64, 0);                        // dg -> [0,16)
    launch_mlp(stream, LOC, 64, ROWS, 32, 32, P(43), 32, nullptr, nullptr, nullptr,
               1.0f, nullptr, 0, ABUF, 64, 0);                       // logits
    reduce_att_kernel<<<(unsigned)((PTS * 32 + 255) / 256), 256, 0, stream>>>(
        ABUF, LOC, 32, SATT, SRAW, PTS);
    launch_mlp(stream, SATT, 64, PTS, 32, 32, P(52), 16, P(53), P(54), P(55),
               0.2f, nullptr, 0, H2, 16, 0);                         // att_out
    launch_mlp(stream, SRAW, 64, PTS, 32, 32, P(56), 64, P(57), P(58), P(59),
               0.2f, nullptr, 0, OUT_REC, 64, 0);                    // rec output

    // ---- LFA2 (ch=32, in C=16) ----
    build_rel_kernel<<<rgrid, 256, 0, stream>>>(GC, IDX, ABUF, NP, ROWS);
    launch_mlp(stream, ABUF, 64, ROWS, 10, 32, P(65), 16, P(66), P(67), P(68),
               0.2f, nullptr, 0, LOC, 64, 48);                       // lse -> [48,64)
    gather_xj_kernel<<<rgrid, 256, 0, stream>>>(H2, 16, 16, IDX, LOC, 32, NP, ROWS);
    build_edge_kernel<<<rgrid, 256, 0, stream>>>(H2, 16, 16, IDX, ABUF, NP, ROWS);
    launch_mlp(stream, ABUF, 64, ROWS, 48, 64, P(61), 32, P(62), P(63), P(64),
               0.2f, nullptr, 0, LOC, 64, 0);                        // dg -> [0,32)
    launch_mlp(stream, LOC, 64, ROWS, 64, 64, P(60), 64, nullptr, nullptr, nullptr,
               1.0f, nullptr, 0, ABUF, 64, 0);                       // logits
    reduce_att_kernel<<<(unsigned)((PTS * 64 + 255) / 256), 256, 0, stream>>>(
        ABUF, LOC, 64, SATT, nullptr, PTS);
    launch_mlp(stream, SATT, 64, PTS, 64, 64, P(69), 32, P(70), P(71), P(72),
               0.2f, nullptr, 0, H3, 32, 0);

    // mlp2 (no act) + shortcut residual + lrelu 0.2  -> f_p output
    launch_mlp(stream, H3, 32, PTS, 32, 32, P(77), 64, P(78), P(79), P(80),
               0.2f, SHORTB, 64, OUT_FP, 64, 0);

    // ======================= voxel branch =======================
    vox_stats_kernel<<<NB, 256, 0, stream>>>(LC, STATS, NP);

    const int scales[3] = { 32, 16, 8 };
    float* gouts[3] = { G32, G16, G8 };
    for (int si = 0; si < 3; si++) {
        int r = scales[si];
        long r3 = (long)r * r * r;
        fill0_kernel<<<(unsigned)((NB * r3 * CIN_ + 255) / 256), 256, 0, stream>>>(
            VSUM, NB * r3 * CIN_);
        fill0_kernel<<<(unsigned)((NB * r3 + 255) / 256), 256, 0, stream>>>(
            VCNT, NB * r3);
        vox_accum_kernel<<<(unsigned)((PTS + 255) / 256), 256, 0, stream>>>(
            LC, Pf, STATS, r, VSUM, VCNT, (si == 0) ? NC0 : nullptr, NP, PTS);
        vox_final_kernel<<<(unsigned)((NB * r3 + 255) / 256), 256, 0, stream>>>(
            VSUM, VCNT, GIN, r, NB * r3);

        long tiles = (long)NB * (r3 / 16) * (COUT_ / 16);
        conv3d_wmma_kernel<1><<<(unsigned)((tiles + 7) / 8), 256, 0, stream>>>(
            GIN, P(3 + 8 * si + 0), P(3 + 8 * si + 1), P(3 + 8 * si + 2),
            P(3 + 8 * si + 3), 0.1f, GTMP, r, COUT_, NB);
        conv3d_wmma_kernel<2><<<(unsigned)((tiles + 7) / 8), 256, 0, stream>>>(
            GTMP, P(3 + 8 * si + 4), P(3 + 8 * si + 5), P(3 + 8 * si + 6),
            P(3 + 8 * si + 7), 0.1f, gouts[si], r, COUT_, NB);
    }

    upgate_kernel<<<(unsigned)((NB * 32768) / 4), 256, 0, stream>>>(
        G32, G16, G8,
        P(31), P(32), P(33), P(34),
        P(35), P(36), P(37), P(38),
        P(39), P(40), P(41), P(42), FIN);

    launch_mlp(stream, FIN, 64, (long)NB * 32768, 64, 64, P(27), 64,
               P(28), P(29), P(30), 0.2f, nullptr, 0, FOUT, 64, 0);

    devox_kernel<<<(unsigned)(PTS / 4), 256, 0, stream>>>(FOUT, NC0, OUT_DVX);
}
